// TDCM_77309411328154
// MI455X (gfx1250) — matrix operand order repaired and checked by reading, not compiled, not run on hardware
//
#include <hip/hip_runtime.h>
#include <hip/hip_bf16.h>

// ---------------------------------------------------------------------------
// B=64, L=1024, E=256, K=8.
//   q = series@Wq.T+bq ; k = text@Wk.T+bk ; v = text@Wv.T+bv (only 16 cols used)
//   mean_value[b,tau] = (1/E) * sum_t <q[b,(t+tau)%L,:], k[b,t,:]>
//     = wrapped-diagonal sums of G_b = Q_b K_b^T, accumulated per block-diagonal
//       entirely inside WMMA C registers (64 tiles -> one 16x16 scatter).
//   Only argmax/argmin of the batch-mean and the per-row softmax denominator
//   are needed (argsort is a red herring); v only needs 16 of 256 channels.
// Memory-bound (~300MB @ 23.3TB/s ~ 15us); bf16 WMMA w/ f32 accumulate.
// ---------------------------------------------------------------------------

typedef __bf16 bfloat;
typedef __attribute__((ext_vector_type(16))) __bf16 v16bf;
typedef __attribute__((ext_vector_type(8)))  __bf16 bf8;
typedef __attribute__((ext_vector_type(8)))  float  v8f;
typedef __attribute__((ext_vector_type(4)))  float  f4;

// gfx1250 native bf16: lowers to v_cvt_pk_bf16_f32 (verified in round-2 disasm)
static __device__ __forceinline__ __bf16 f2bf(float f) { return (__bf16)f; }

#define Ec 256
#define Lc 1024
#define Bc 64

// ---------------- elementwise f32 -> bf16 (weights, done once) --------------
__global__ void tdcm_cvt_bf16(const float* __restrict__ src, bfloat* __restrict__ dst, int n) {
  int t = blockIdx.x * blockDim.x + threadIdx.x;
  if (t < n) dst[t] = f2bf(src[t]);
}

// ---------------- zero scratch accumulator (graph-replay determinism) -------
__global__ void tdcm_zero_f32(float* p, int n) {
  int t = blockIdx.x * blockDim.x + threadIdx.x;
  if (t < n) p[t] = 0.0f;
}

// ---------------- Q/K GEMM: one wave per 16x128 output strip ----------------
// 8 WMMA accumulators (64 VGPRs) -> no spills; A converted once per strip with
// v_cvt_pk_bf16_f32; B is pre-converted bf16 weights (pure b128 loads).
// A layout (16-bit 16x32): lane l: row M=l&15; elems 0..7 at K=8*(l>>4)+0..7,
// elems 8..15 at K=16+8*(l>>4)+0..7. B (32x16): lane l: col N=l&15, same K order as A.
__global__ void tdcm_gemm_qk(const float* __restrict__ S, const float* __restrict__ T,
                             const bfloat* __restrict__ Wqb, const float* __restrict__ bq,
                             const bfloat* __restrict__ Wkb, const float* __restrict__ bk,
                             bfloat* __restrict__ qout, bfloat* __restrict__ kout) {
  const float* X; const bfloat* W; const float* bias; bfloat* out;
  if (blockIdx.z == 0) { X = S; W = Wqb; bias = bq; out = qout; }
  else                 { X = T; W = Wkb; bias = bk; out = kout; }
  const int row0 = blockIdx.x * 16;
  const int ct0  = blockIdx.y * 8;               // column-tile half: tiles ct0..ct0+7
  const int lane = threadIdx.x;
  const int m = lane & 15, half = lane >> 4;

  v8f acc[8];
#pragma unroll
  for (int ct = 0; ct < 8; ++ct) acc[ct] = (v8f){};

  const float* arow = X + (size_t)(row0 + m) * Ec;
  for (int kk = 0; kk < Ec; kk += 32) {
    const float* ap = arow + kk + 8 * half;
    f4 a0 = *(const f4*)(ap);      f4 a1 = *(const f4*)(ap + 4);
    f4 a2 = *(const f4*)(ap + 16); f4 a3 = *(const f4*)(ap + 20);
    v16bf A;
#pragma unroll
    for (int i = 0; i < 4; ++i) {
      A[i] = f2bf(a0[i]); A[4 + i] = f2bf(a1[i]);
      A[8 + i] = f2bf(a2[i]); A[12 + i] = f2bf(a3[i]);
    }
#pragma unroll
    for (int ct = 0; ct < 8; ++ct) {
      const bfloat* bp = W + (size_t)((ct0 + ct) * 16 + m) * Ec + kk + 8 * half;
      bf8 b0 = *(const bf8*)(bp);
      bf8 b1 = *(const bf8*)(bp + 16);
      v16bf Bm;
#pragma unroll
      for (int i = 0; i < 8; ++i) { Bm[i] = b0[i]; Bm[8 + i] = b1[i]; }
      acc[ct] = __builtin_amdgcn_wmma_f32_16x16x32_bf16(false, A, false, Bm, (short)0, acc[ct], false, false);
    }
  }
#pragma unroll
  for (int ct = 0; ct < 8; ++ct) {
    const float bb = bias[(ct0 + ct) * 16 + m];
#pragma unroll
    for (int r = 0; r < 8; ++r) {               // C: M = 8*half + r, N = m
      const int row = row0 + 8 * half + r;
      out[(size_t)row * Ec + (ct0 + ct) * 16 + m] = f2bf(acc[ct][r] + bb);
    }
  }
}

// ---------------- V partial GEMM: channels 0..7 and 248..255 ----------------
__global__ void tdcm_gemm_v(const float* __restrict__ T, const bfloat* __restrict__ Wvb,
                            const float* __restrict__ bv, float* __restrict__ vout) {
  const int row0 = blockIdx.x * 16;
  const int lane = threadIdx.x;
  const int m = lane & 15, half = lane >> 4;
  const int chan = (m < 8) ? m : (240 + m);     // 0..7, 248..255

  v8f c = {};
  const float* arow = T + (size_t)(row0 + m) * Ec;
  const bfloat* brow = Wvb + (size_t)chan * Ec;
  for (int kk = 0; kk < Ec; kk += 32) {
    const float* ap = arow + kk + 8 * half;
    f4 a0 = *(const f4*)(ap);      f4 a1 = *(const f4*)(ap + 4);
    f4 a2 = *(const f4*)(ap + 16); f4 a3 = *(const f4*)(ap + 20);
    bf8 b0 = *(const bf8*)(brow + kk + 8 * half);
    bf8 b1 = *(const bf8*)(brow + kk + 16 + 8 * half);
    v16bf A, Bm;
#pragma unroll
    for (int i = 0; i < 4; ++i) {
      A[i] = f2bf(a0[i]); A[4 + i] = f2bf(a1[i]);
      A[8 + i] = f2bf(a2[i]); A[12 + i] = f2bf(a3[i]);
    }
#pragma unroll
    for (int i = 0; i < 8; ++i) { Bm[i] = b0[i]; Bm[8 + i] = b1[i]; }
    c = __builtin_amdgcn_wmma_f32_16x16x32_bf16(false, A, false, Bm, (short)0, c, false, false);
  }
  const float bb = bv[chan];
#pragma unroll
  for (int r = 0; r < 8; ++r)
    vout[(size_t)(row0 + 8 * half + r) * 16 + m] = c[r] + bb;
}

// ---------------- correlation: block-diagonal accumulation of G = Q K^T -----
// 4 waves per workgroup handle deltas d0..d0+3 for one batch b; the shared
// k-tile for each cb is staged into double-buffered LDS with CDNA5 async
// loads (global_load_async_to_lds_b128, ASYNCcnt), giving 4x k-side reuse.
// Element (M,j) of the accumulated tile scatters to tau=(16*delta+M-j)%1024.
__global__ void tdcm_autocorr(const bfloat* __restrict__ q, const bfloat* __restrict__ k,
                              float* __restrict__ raw) {
  extern __shared__ char smem[];                 // 16 KB = 2 x 8 KB k-tiles
  bfloat* tile = (bfloat*)smem;
  const int b    = blockIdx.y;
  const int t    = threadIdx.x;                  // 0..127
  const int wave = t >> 5;
  const int lane = t & 31;
  const int delta = blockIdx.x * 4 + wave;
  const int m = lane & 15, half = lane >> 4;

  const bfloat* qb = q + (size_t)b * Lc * Ec;
  const unsigned long long kbase =
      (unsigned long long)(const void*)(k + (size_t)b * Lc * Ec);
  const unsigned ldsbase = __builtin_amdgcn_groupstaticsize();

  // stage k-tile for cb=0 into buffer 0 (128 lanes x 4 x 16B = 8 KB)
#pragma unroll
  for (int r2 = 0; r2 < 4; ++r2) {
    unsigned byteoff = (unsigned)((r2 * 128 + t) * 16);
    unsigned lds  = ldsbase + byteoff;
    unsigned voff = byteoff;
    asm volatile("global_load_async_to_lds_b128 %0, %1, %2"
                 :: "v"(lds), "v"(voff), "s"(kbase) : "memory");
  }
  asm volatile("s_wait_asynccnt 0x0" ::: "memory");
  __syncthreads();

  v8f c = {};
  for (int cb = 0; cb < 64; ++cb) {
    // kick off async stage of next k-tile into the other buffer
    if (cb + 1 < 64) {
      unsigned bufoff = (unsigned)(((cb + 1) & 1) * 8192);
#pragma unroll
      for (int r2 = 0; r2 < 4; ++r2) {
        unsigned byteoff = (unsigned)((r2 * 128 + t) * 16);
        unsigned lds  = ldsbase + bufoff + byteoff;
        unsigned voff = (unsigned)((cb + 1) * 8192) + byteoff;
        asm volatile("global_load_async_to_lds_b128 %0, %1, %2"
                     :: "v"(lds), "v"(voff), "s"(kbase) : "memory");
      }
    }
    const int rb = (cb + delta) & 63;
    const bfloat* arow = qb + (size_t)(rb * 16 + m) * Ec;
    if (cb + 1 < 64) {                           // gfx1250 global_prefetch_b8
      const int rb2 = (cb + 1 + delta) & 63;
      __builtin_prefetch(qb + (size_t)(rb2 * 16 + m) * Ec, 0, 0);
    }
    const bfloat* brow = tile + (cb & 1) * 4096 + m * Ec;
#pragma unroll
    for (int kk = 0; kk < Ec; kk += 32) {
      bf8 a0 = *(const bf8*)(arow + kk + 8 * half);
      bf8 a1 = *(const bf8*)(arow + kk + 16 + 8 * half);
      bf8 b0 = *(const bf8*)(brow + kk + 8 * half);
      bf8 b1 = *(const bf8*)(brow + kk + 16 + 8 * half);
      v16bf A, Bm;
#pragma unroll
      for (int i = 0; i < 8; ++i) {
        A[i] = a0[i]; A[8 + i] = a1[i];
        Bm[i] = b0[i]; Bm[8 + i] = b1[i];
      }
      c = __builtin_amdgcn_wmma_f32_16x16x32_bf16(false, A, false, Bm, (short)0, c, false, false);
    }
    // next-tile async writes must land + everyone done reading before swap
    asm volatile("s_wait_asynccnt 0x0" ::: "memory");
    __syncthreads();
  }
#pragma unroll
  for (int r = 0; r < 8; ++r) {
    const int M = 8 * half + r;
    const int tau = (16 * delta + M - m) & (Lc - 1);
    atomicAdd(&raw[(size_t)b * Lc + tau], c[r]);
  }
}

// ---------------- argmax/argmin of column sums ------------------------------
__global__ void tdcm_reduce_cols(const float* __restrict__ raw, int* __restrict__ idx2) {
  const int tau = threadIdx.x;                  // 1024 threads
  float s = 0.0f;
  for (int b = 0; b < Bc; ++b) s += raw[(size_t)b * Lc + tau];
  __shared__ float sv[1024]; __shared__ int si[1024];
  __shared__ float mv[1024]; __shared__ int mi[1024];
  sv[tau] = s; si[tau] = tau; mv[tau] = s; mi[tau] = tau;
  __syncthreads();
  for (int off = 512; off > 0; off >>= 1) {
    if (tau < off) {
      if (sv[tau + off] > sv[tau] || (sv[tau + off] == sv[tau] && si[tau + off] < si[tau]))
        { sv[tau] = sv[tau + off]; si[tau] = si[tau + off]; }
      if (mv[tau + off] < mv[tau] || (mv[tau + off] == mv[tau] && mi[tau + off] > mi[tau]))
        { mv[tau] = mv[tau + off]; mi[tau] = mi[tau + off]; }
    }
    __syncthreads();
  }
  if (tau == 0) { idx2[0] = si[0]; idx2[1] = mi[0]; }
}

// ---------------- per-batch softmax weights at imax/imin --------------------
__global__ void tdcm_softmax_w(const float* __restrict__ raw, const int* __restrict__ idx2,
                               float* __restrict__ w12) {
  const int b = blockIdx.x, t = threadIdx.x;    // 256 threads
  const float* row = raw + (size_t)b * Lc;
  const float s = 1.0f / (16.0f * 256.0f);      // scale(1/16) * mean(1/E)
  float m = -3.4e38f;
  for (int j = t; j < Lc; j += 256) m = fmaxf(m, row[j]);
  __shared__ float red[256];
  red[t] = m; __syncthreads();
  for (int off = 128; off > 0; off >>= 1) { if (t < off) red[t] = fmaxf(red[t], red[t + off]); __syncthreads(); }
  m = red[0]; __syncthreads();
  float e = 0.0f;
  for (int j = t; j < Lc; j += 256) e += __expf((row[j] - m) * s);
  red[t] = e; __syncthreads();
  for (int off = 128; off > 0; off >>= 1) { if (t < off) red[t] += red[t + off]; __syncthreads(); }
  if (t == 0) {
    const float d = red[0];
    w12[b]      = __expf((row[idx2[0]] - m) * s) / d;
    w12[Bc + b] = __expf((row[idx2[1]] - m) * s) / d;
  }
}

// ---------------- gather rolled v-channels * weights ------------------------
__global__ void tdcm_gather(const float* __restrict__ v16, const int* __restrict__ idx2,
                            const float* __restrict__ w12, float* __restrict__ out) {
  const int t = blockIdx.x * blockDim.x + threadIdx.x;  // 65536 = B*L
  const int b = t >> 10, l = t & (Lc - 1);
  const int r1 = (l + idx2[0]) & (Lc - 1);
  const int r2 = (l + idx2[1]) & (Lc - 1);
  const float w1 = w12[b], w2 = w12[Bc + b];
  const f4* p1 = (const f4*)(v16 + (size_t)(b * Lc + r1) * 16);      // chans 0..7
  const f4* p2 = (const f4*)(v16 + (size_t)(b * Lc + r2) * 16 + 8);  // chans 248..255
  f4* o1 = (f4*)(out + (size_t)t * 8);
  f4* o2 = (f4*)(out + (size_t)Bc * Lc * 8 + (size_t)t * 8);
  o1[0] = p1[0] * w1; o1[1] = p1[1] * w1;
  o2[0] = p2[0] * w2; o2[1] = p2[1] * w2;
}

// ---------------------------------------------------------------------------
extern "C" void kernel_launch(void* const* d_in, const int* in_sizes, int n_in,
                              void* d_out, int out_size, void* d_ws, size_t ws_size,
                              hipStream_t stream) {
  const float* series = (const float*)d_in[0];
  const float* text   = (const float*)d_in[1];
  const float* Wq     = (const float*)d_in[2];
  const float* bq     = (const float*)d_in[3];
  const float* Wk     = (const float*)d_in[4];
  const float* bk     = (const float*)d_in[5];
  const float* Wv     = (const float*)d_in[6];
  const float* bv     = (const float*)d_in[7];
  float* out = (float*)d_out;

  char* ws = (char*)d_ws;
  // workspace layout (bytes), total ~68.6 MB
  const size_t OFF_Q   = 0;                       // 32 MB bf16 q
  const size_t OFF_K   = 33554432;                // 32 MB bf16 k
  const size_t OFF_V   = 67108864;                // 4 MB f32 v16
  const size_t OFF_RAW = 71303168;                // 256 KB f32 accum
  const size_t OFF_IDX = 71565312;                // 2 ints
  const size_t OFF_W   = 71565328;                // 128 f32 weights
  const size_t OFF_WQB = 71566336;                // 128 KB bf16 Wq
  const size_t OFF_WKB = 71697408;                // 128 KB bf16 Wk
  const size_t OFF_WVB = 71828480;                // 128 KB bf16 Wv
  bfloat* qbf  = (bfloat*)(ws + OFF_Q);
  bfloat* kbf  = (bfloat*)(ws + OFF_K);
  float*  v16  = (float*)(ws + OFF_V);
  float*  raw  = (float*)(ws + OFF_RAW);
  int*    idx2 = (int*)(ws + OFF_IDX);
  float*  w12  = (float*)(ws + OFF_W);
  bfloat* wqb  = (bfloat*)(ws + OFF_WQB);
  bfloat* wkb  = (bfloat*)(ws + OFF_WKB);
  bfloat* wvb  = (bfloat*)(ws + OFF_WVB);

  // one-time weight conversions (tiny) + accumulator zero
  tdcm_cvt_bf16<<<Ec * Ec / 256, 256, 0, stream>>>(Wq, wqb, Ec * Ec);
  tdcm_cvt_bf16<<<Ec * Ec / 256, 256, 0, stream>>>(Wk, wkb, Ec * Ec);
  tdcm_cvt_bf16<<<Ec * Ec / 256, 256, 0, stream>>>(Wv, wvb, Ec * Ec);
  tdcm_zero_f32<<<(Bc * Lc) / 256, 256, 0, stream>>>(raw, Bc * Lc);

  dim3 gqk((Bc * Lc) / 16, 2, 2);                 // rows x col-half x {q,k}
  tdcm_gemm_qk<<<gqk, 32, 0, stream>>>(series, text, wqb, bq, wkb, bk, qbf, kbf);
  tdcm_gemm_v<<<(Bc * Lc) / 16, 32, 0, stream>>>(text, wvb, bv, v16);

  dim3 gac(16, Bc);                               // 4 deltas per workgroup
  tdcm_autocorr<<<gac, 128, 16384, stream>>>(qbf, kbf, raw);

  tdcm_reduce_cols<<<1, 1024, 0, stream>>>(raw, idx2);
  tdcm_softmax_w<<<Bc, 256, 0, stream>>>(raw, idx2, w12);
  tdcm_gather<<<(Bc * Lc) / 256, 256, 0, stream>>>(v16, idx2, w12, out);
}
